// WindowAttention_29772713296052
// MI455X (gfx1250) — matrix operand order repaired and checked by reading, not compiled, not run on hardware
//
#include <hip/hip_runtime.h>
#include <hip/hip_bf16.h>
#include <math.h>

typedef __attribute__((ext_vector_type(16))) _Float16   v16h;
typedef __attribute__((ext_vector_type(8)))  float      v8f;
typedef __attribute__((ext_vector_type(4)))  unsigned   u32x4;
typedef __attribute__((ext_vector_type(4)))  float      f32x4;

#define WAVES 8
#define BLOCK 256

// ---- LDS layout (dynamic shared memory), all offsets 16B aligned ----
// Xs region is reused: phase0/1 = X (f16 64x264), phase2 = scores (f32 64x68),
// phase3/4 = attention output O (f16 64x264).
#define XS_OFF   0
#define QS_OFF   33792
#define KS_OFF   67584
#define VT_OFF   101376   // V transposed: f16 256 x 72
#define PS_OFF   138240   // softmax probs: f16 64 x 72
#define SMEM_BYTES 147456

#define XS_STRIDE 264     // 256 + 8 pad (multiple of 8 for b128, breaks bank conflicts)
#define VT_STRIDE 72
#define PS_STRIDE 72
#define AS_STRIDE 68      // f32 scores stride

// ---- workspace layout ----
#define WTQ_OFF   0        // f16 [768][256]  (qkv_w transposed)
#define PWT_OFF   393216   // f16 [256][256]  (proj_w transposed)
#define BIAS_OFF  524288   // f32 [64][64][64] combined phi+theta bias

// A-fragment: 16x32 f16 tile, row m = lane&15.
// lanes 0-15: K = 0..7 then 16..23 ; lanes 16-31: K = 8..15 then 24..31
__device__ inline v16h load_frag_a(const _Float16* base, int row0, int stride,
                                   int k0, int lane) {
  int half = lane >> 4;
  const _Float16* p = base + (row0 + (lane & 15)) * stride + k0 + half * 8;
  union { v16h v; u32x4 q[2]; } t;
  t.q[0] = *(const u32x4*)p;
  t.q[1] = *(const u32x4*)(p + 16);
  return t.v;
}

// B-fragment: 32x16 f16 tile (KxN), col n = lane&15.
// lanes 0-15: K = 0..7 then 16..23 ; lanes 16-31: K = 8..15 then 24..31
__device__ inline v16h load_frag_b(const _Float16* base, int col0, int stride,
                                   int k0, int lane) {
  int half = lane >> 4;
  const _Float16* p = base + (col0 + (lane & 15)) * stride + k0 + half * 8;
  union { v16h v; u32x4 q[2]; } t;
  t.q[0] = *(const u32x4*)p;
  t.q[1] = *(const u32x4*)(p + 16);
  return t.v;
}

__device__ inline v8f wmma16(v16h a, v16h b, v8f c) {
  return __builtin_amdgcn_wmma_f32_16x16x32_f16(false, a, false, b,
                                                (short)0, c, false, false);
}

// ---------------- prep kernel 1: convert + transpose weights to f16 --------
__global__ void prep_weights(const float* __restrict__ qkv_w,
                             const float* __restrict__ proj_w,
                             _Float16* __restrict__ wtq,
                             _Float16* __restrict__ pwt) {
  int id = blockIdx.x * BLOCK + threadIdx.x;      // grid = 1024 blocks
  if (id < 768 * 256) {
    int n = id >> 8, k = id & 255;                // wtq[n][k] = qkv_w[k][n]
    wtq[id] = (_Float16)qkv_w[k * 768 + n];
  } else {
    int id2 = id - 768 * 256;
    int n = id2 >> 8, k = id2 & 255;              // pwt[n][k] = proj_w[k][n]
    pwt[id2] = (_Float16)proj_w[k * 256 + n];
  }
}

// ---------------- prep kernel 2: combined bias table [theta][i][j] ---------
__global__ void prep_bias(const float* __restrict__ theta_max,
                          const float* __restrict__ a_p, const float* __restrict__ b_p,
                          const float* __restrict__ a_r, const float* __restrict__ b_r,
                          float* __restrict__ bias) {
  int b8 = blockIdx.x;                            // grid = 64 blocks (theta index)
  float tm = theta_max[b8];
  for (int p = threadIdx.x; p < 4096; p += BLOCK) {
    int i = p >> 6, j = p & 63;
    int xi = i & 7, yi = i >> 3, xj = j & 7, yj = j >> 3;
    int az  = xi - xj;                            // azimuth (cols), [-7,7]
    int rad = yi - yj;                            // radius  (rows), [-7,7]
    int azi = az  < 0 ? az  + 15 : az;            // np.mod(., 2*Ww-1)
    int ri  = rad < 0 ? rad + 15 : rad;
    float aza = (float)az * (2.0f * 3.14159265358979f / 64.0f);
    float aphi = a_p[azi] * __cosf(aza) + b_p[azi] * __sinf(aza);
    float rn = (float)rad * (tm * (1.0f / 64.0f));
    float ath = a_r[ri] * __cosf(rn) + b_r[ri] * __sinf(rn);
    bias[b8 * 4096 + p] = aphi + ath;
  }
}

// ---------------- fused attention: one workgroup per window ----------------
__global__ void __launch_bounds__(BLOCK)
fused_attn_kernel(const float* __restrict__ xg,
                  const float* __restrict__ qkv_b,
                  const float* __restrict__ proj_b,
                  const _Float16* __restrict__ wtq,   // [768][256]
                  const _Float16* __restrict__ pwt,   // [256][256]
                  const float* __restrict__ bias,     // [64][4096]
                  float* __restrict__ out) {
  extern __shared__ __align__(16) char smem[];
  _Float16* Xs = (_Float16*)(smem + XS_OFF);
  _Float16* Qs = (_Float16*)(smem + QS_OFF);
  _Float16* Ks = (_Float16*)(smem + KS_OFF);
  _Float16* Vt = (_Float16*)(smem + VT_OFF);
  _Float16* Ps = (_Float16*)(smem + PS_OFF);

  const int tid = threadIdx.x, lane = tid & 31, wave = tid >> 5;
  const int gid = blockIdx.x;
  const int mt  = wave & 3;                  // this wave's fixed 16-row stripe
  const int m0  = mt * 16;
  const int ntA = wave >> 2;                 // first column-tile, step 2
  const int Nc  = lane & 15;
  const int Mo  = m0 + ((lane >> 4) << 3);

  // ---- prefetch attention bias for this wave's 2 score tiles (hide L2 lat) --
  const float* bias_g = bias + (size_t)(gid >> 6) * 4096;
  float bpre[2][8];
  #pragma unroll
  for (int s2 = 0; s2 < 2; ++s2) {
    int n0 = (ntA + 2 * s2) * 16;
    #pragma unroll
    for (int v = 0; v < 8; ++v)
      bpre[s2][v] = bias_g[(Mo + v) * 64 + n0 + Nc];
  }

  // ---- phase 0: stream x tile (64x256 f32) in, convert to f16 in LDS ----
  const f32x4* xv = (const f32x4*)(xg + (size_t)gid * (64 * 256));
  #pragma unroll
  for (int i = 0; i < 16; ++i) {
    int idx = tid + BLOCK * i;                 // 4096 float4 total
    f32x4 f = xv[idx];
    int r = idx >> 6, c = (idx & 63) << 2;
    _Float16* d = Xs + r * XS_STRIDE + c;
    d[0] = (_Float16)f.x; d[1] = (_Float16)f.y;
    d[2] = (_Float16)f.z; d[3] = (_Float16)f.w;
  }
  __syncthreads();

  // ---- phase 1: QKV = X @ Wqkv + b  (M=64, N=768, K=256) ----
  {
    v16h ax[8];                                // hoist X stripe: 16x256 in regs
    #pragma unroll
    for (int kc = 0; kc < 8; ++kc)
      ax[kc] = load_frag_a(Xs, m0, XS_STRIDE, kc * 32, lane);
    for (int nt = ntA; nt < 48; nt += 2) {     // 24 column tiles per wave
      int n0 = nt * 16;
      v8f acc = {0.f, 0.f, 0.f, 0.f, 0.f, 0.f, 0.f, 0.f};
      #pragma unroll
      for (int kc = 0; kc < 8; ++kc)
        acc = wmma16(ax[kc], load_frag_b(wtq, n0, 256, kc * 32, lane), acc);
      int N = n0 + Nc;
      float bv = qkv_b[N];
      if (n0 < 256) {                 // Q, pre-scaled by DIM^-0.5 = 1/16
        #pragma unroll
        for (int v = 0; v < 8; ++v)
          Qs[(Mo + v) * XS_STRIDE + N] = (_Float16)((acc[v] + bv) * 0.0625f);
      } else if (n0 < 512) {          // K (token-major -> rows contiguous in K)
        #pragma unroll
        for (int v = 0; v < 8; ++v)
          Ks[(Mo + v) * XS_STRIDE + (N - 256)] = (_Float16)(acc[v] + bv);
      } else {                        // V stored transposed for AV B-fragments
        #pragma unroll
        for (int v = 0; v < 8; ++v)
          Vt[(N - 512) * VT_STRIDE + Mo + v] = (_Float16)(acc[v] + bv);
      }
    }
  }
  __syncthreads();

  // ---- phase 2: S = Q @ K^T + bias(theta,i,j)  (64x64, K=256) ----
  float* As = (float*)(smem + XS_OFF);                  // reuse X region
  {
    v16h aq[8];                                // hoist Q stripe
    #pragma unroll
    for (int kc = 0; kc < 8; ++kc)
      aq[kc] = load_frag_a(Qs, m0, XS_STRIDE, kc * 32, lane);
    #pragma unroll
    for (int s2 = 0; s2 < 2; ++s2) {
      int n0 = (ntA + 2 * s2) * 16;
      v8f acc = {0.f, 0.f, 0.f, 0.f, 0.f, 0.f, 0.f, 0.f};
      #pragma unroll
      for (int kc = 0; kc < 8; ++kc)
        acc = wmma16(aq[kc], load_frag_b(Ks, n0, XS_STRIDE, kc * 32, lane), acc);
      int N = n0 + Nc;
      #pragma unroll
      for (int v = 0; v < 8; ++v)
        As[(Mo + v) * AS_STRIDE + N] = acc[v] + bpre[s2][v];
    }
  }
  __syncthreads();

  // ---- softmax: 4 threads per row, shuffle-combined ----
  {
    int r = tid >> 2, q = tid & 3;             // row, 16-col quarter
    const float* rowp = As + r * AS_STRIDE + q * 16;
    float e[16];
    float mx = -1e30f;
    #pragma unroll
    for (int j = 0; j < 16; ++j) { e[j] = rowp[j]; mx = fmaxf(mx, e[j]); }
    mx = fmaxf(mx, __shfl_xor(mx, 1));
    mx = fmaxf(mx, __shfl_xor(mx, 2));
    float s = 0.f;
    #pragma unroll
    for (int j = 0; j < 16; ++j) { e[j] = __expf(e[j] - mx); s += e[j]; }
    s += __shfl_xor(s, 1);
    s += __shfl_xor(s, 2);
    float inv = 1.f / s;
    union { _Float16 h[16]; u32x4 q4[2]; } ob;
    #pragma unroll
    for (int j = 0; j < 16; ++j) ob.h[j] = (_Float16)(e[j] * inv);
    u32x4* pp = (u32x4*)(Ps + r * PS_STRIDE + q * 16);
    pp[0] = ob.q4[0];
    pp[1] = ob.q4[1];
  }
  __syncthreads();

  // ---- phase 3: O = P @ V  (64x256, K=64), V pre-transposed ----
  _Float16* Os = (_Float16*)(smem + XS_OFF);            // reuse X/scores region
  {
    v16h ap[2];
    ap[0] = load_frag_a(Ps, m0, PS_STRIDE, 0, lane);
    ap[1] = load_frag_a(Ps, m0, PS_STRIDE, 32, lane);
    for (int nt = ntA; nt < 16; nt += 2) {     // 8 column tiles per wave
      int n0 = nt * 16;
      v8f acc = {0.f, 0.f, 0.f, 0.f, 0.f, 0.f, 0.f, 0.f};
      acc = wmma16(ap[0], load_frag_b(Vt, n0, VT_STRIDE, 0, lane), acc);
      acc = wmma16(ap[1], load_frag_b(Vt, n0, VT_STRIDE, 32, lane), acc);
      int N = n0 + Nc;
      #pragma unroll
      for (int v = 0; v < 8; ++v)
        Os[(Mo + v) * XS_STRIDE + N] = (_Float16)acc[v];
    }
  }
  __syncthreads();

  // ---- phase 4: out = O @ Wproj + b, stream f32 to HBM ----
  float* og = out + (size_t)gid * (64 * 256);
  {
    v16h ao[8];                                // hoist O stripe
    #pragma unroll
    for (int kc = 0; kc < 8; ++kc)
      ao[kc] = load_frag_a(Os, m0, XS_STRIDE, kc * 32, lane);
    for (int nt = ntA; nt < 16; nt += 2) {
      int n0 = nt * 16;
      v8f acc = {0.f, 0.f, 0.f, 0.f, 0.f, 0.f, 0.f, 0.f};
      #pragma unroll
      for (int kc = 0; kc < 8; ++kc)
        acc = wmma16(ao[kc], load_frag_b(pwt, n0, 256, kc * 32, lane), acc);
      int N = n0 + Nc;
      float bv = proj_b[N];
      #pragma unroll
      for (int v = 0; v < 8; ++v)
        og[(Mo + v) * 256 + N] = acc[v] + bv;
    }
  }
}

extern "C" void kernel_launch(void* const* d_in, const int* in_sizes, int n_in,
                              void* d_out, int out_size, void* d_ws, size_t ws_size,
                              hipStream_t stream) {
  (void)in_sizes; (void)n_in; (void)out_size; (void)ws_size;
  const float* x         = (const float*)d_in[0];
  const float* theta_max = (const float*)d_in[1];
  const float* qkv_w     = (const float*)d_in[2];
  const float* qkv_b     = (const float*)d_in[3];
  const float* proj_w    = (const float*)d_in[4];
  const float* proj_b    = (const float*)d_in[5];
  const float* a_p       = (const float*)d_in[6];
  const float* b_p       = (const float*)d_in[7];
  const float* a_r       = (const float*)d_in[8];
  const float* b_r       = (const float*)d_in[9];

  char* ws = (char*)d_ws;
  _Float16* wtq = (_Float16*)(ws + WTQ_OFF);
  _Float16* pwt = (_Float16*)(ws + PWT_OFF);
  float*    bias = (float*)(ws + BIAS_OFF);

  prep_weights<<<1024, BLOCK, 0, stream>>>(qkv_w, proj_w, wtq, pwt);
  prep_bias<<<64, BLOCK, 0, stream>>>(theta_max, a_p, b_p, a_r, b_r, bias);

  hipFuncSetAttribute((const void*)fused_attn_kernel,
                      hipFuncAttributeMaxDynamicSharedMemorySize, SMEM_BYTES);
  fused_attn_kernel<<<4096, BLOCK, SMEM_BYTES, stream>>>(
      x, qkv_b, proj_b, wtq, pwt, bias, (float*)d_out);
}
